// DAttentionBaseline_3289944949130
// MI455X (gfx1250) — matrix operand order repaired and checked by reading, not compiled, not run on hardware
//
#include <hip/hip_runtime.h>
#include <hip/hip_bf16.h>

// ---------------- CDNA5 WMMA types ----------------
typedef __attribute__((ext_vector_type(16))) _Float16 v16h;
typedef __attribute__((ext_vector_type(8)))  float    v8f;

// Problem constants
#define CB   16      // batch
#define CNC  512     // channels
#define CH   32
#define CW   32
#define CHW  1024
#define CNH  8
#define CHC  64
#define CG   4
#define CGC  128
#define CN   256     // Hk*Wk
#define MCHUNK 128   // attention M chunk
#define LDH  34      // LDS row stride (halves), padded

// ---------------- WMMA fragment loaders (ISA 7.12.2 layouts) ----------------
// A 16x32 f16: lanes 0-15 row M=lane, K {0..7}U{16..23}; lanes 16-31 row M=lane-16,
// K {8..15}U{24..31}; 2 halves per VGPR, K ascending within pair.
__device__ __forceinline__ v16h frag_a_from_lds(const _Float16* base) {
  int lane = threadIdx.x & 31;
  int r = lane & 15;
  int kh = (lane >> 4) * 8;            // 0 or 8
  const _Float16* p = base + r * LDH;
  v16h f;
#pragma unroll
  for (int v = 0; v < 8; ++v) {
    int kb = ((v < 4) ? 0 : 16) + kh + (v & 3) * 2;
    f[2 * v]     = p[kb];
    f[2 * v + 1] = p[kb + 1];
  }
  return f;
}

// B 32x16 f16 stored in LDS as [n][k]: lanes 0-15 col N=lane K {0..7}U{16..23};
// lanes 16-31 col N=lane-16 K {8..15}U{24..31}.
__device__ __forceinline__ v16h frag_b_from_lds(const _Float16* base) {
  int lane = threadIdx.x & 31;
  int nn = lane & 15;
  int kh = (lane >> 4) * 8;            // 0 or 8
  const _Float16* p = base + nn * LDH;
  v16h f;
#pragma unroll
  for (int v = 0; v < 8; ++v) {
    int kb = ((v < 4) ? 0 : 16) + kh + (v & 3) * 2;
    f[2 * v]     = p[kb];
    f[2 * v + 1] = p[kb + 1];
  }
  return f;
}

// C layout: VGPR r -> M = r + 8*(lane>=16), N = lane&15.

// ---------------- 1x1 conv projection GEMM (NCHW) ----------------
// Y[b, o, hw] = sum_c X[b, c, hw] * W[o, c] + bias[o]
// 256 threads = 8 waves; block tile 64(M) x 64(N); wave = 16x32 (2 WMMA); K step 32.
__global__ __launch_bounds__(256)
void gemm_proj_kernel(const float* __restrict__ X, const float* __restrict__ W,
                      const float* __restrict__ bias, float* __restrict__ Y,
                      int HW, int Cin, int Cout) {
  __shared__ _Float16 As[64][LDH];
  __shared__ _Float16 Bs[64][LDH];
  int tid = threadIdx.x;
  int wv  = tid >> 5;
  int mi  = (wv & 3) * 16;             // wave M subtile
  int nj  = (wv >> 2) * 32;            // wave N subtile pair
  int m0  = blockIdx.x * 64;
  int n0  = blockIdx.y * 64;
  int b   = m0 / HW;                   // 64 | HW -> tile never straddles a batch
  int hw0 = m0 % HW;
  const float* Xb = X + (size_t)b * Cin * HW + hw0;
  v8f acc0 = {}, acc1 = {};
  for (int k0 = 0; k0 < Cin; k0 += 32) {
    __syncthreads();
    // A: 64 m x 32 k, coalesced along m (contiguous hw)
#pragma unroll
    for (int it = 0; it < 8; ++it) {
      int idx = tid + it * 256;
      int kk = idx >> 6, m = idx & 63;
      As[m][kk] = (_Float16)Xb[(size_t)(k0 + kk) * HW + m];
    }
    // B: 64 n x 32 k, vectorized along k (weights row-major in c)
#pragma unroll
    for (int it = 0; it < 2; ++it) {
      int idx = tid + it * 256;
      int k4 = (idx & 7) * 4, nn = idx >> 3;
      const float4 w4 = *(const float4*)&W[(size_t)(n0 + nn) * Cin + k0 + k4];
      Bs[nn][k4 + 0] = (_Float16)w4.x;
      Bs[nn][k4 + 1] = (_Float16)w4.y;
      Bs[nn][k4 + 2] = (_Float16)w4.z;
      Bs[nn][k4 + 3] = (_Float16)w4.w;
    }
    __syncthreads();
    v16h a  = frag_a_from_lds(&As[mi][0]);
    v16h b0 = frag_b_from_lds(&Bs[nj][0]);
    v16h b1 = frag_b_from_lds(&Bs[nj + 16][0]);
    acc0 = __builtin_amdgcn_wmma_f32_16x16x32_f16(false, a, false, b0, (short)0, acc0,
                                                  false, false);
    acc1 = __builtin_amdgcn_wmma_f32_16x16x32_f16(false, a, false, b1, (short)0, acc1,
                                                  false, false);
  }
  int lane = tid & 31;
  int half = lane >> 4, nn = lane & 15;
  int o0 = n0 + nj + nn, o1 = o0 + 16;
  float bs0 = bias[o0], bs1 = bias[o1];
#pragma unroll
  for (int r = 0; r < 8; ++r) {
    int hw = hw0 + mi + r + 8 * half;
    Y[((size_t)b * Cout + o0) * HW + hw] = acc0[r] + bs0;
    Y[((size_t)b * Cout + o1) * HW + hw] = acc1[r] + bs1;
  }
}

// ---------------- attention scores: S[bh, ml, j] = sum_c q[bh,c,m]*k[bh,c,j] ---------
// Block tile 64(M) x 64(J); K = 64 (2 steps).
__global__ __launch_bounds__(256)
void attn_score_kernel(const float* __restrict__ Q, const float* __restrict__ K,
                       float* __restrict__ S, int m0chunk) {
  __shared__ _Float16 As[64][LDH];
  __shared__ _Float16 Bs[64][LDH];
  int tid = threadIdx.x;
  int wv  = tid >> 5;
  int mi  = (wv & 3) * 16;
  int nj  = (wv >> 2) * 32;
  int bh  = blockIdx.z;
  int mbase = m0chunk + blockIdx.x * 64;
  int n0  = blockIdx.y * 64;
  const float* Qb = Q + (size_t)bh * CHC * CHW + mbase;
  const float* Kb = K + (size_t)bh * CHC * CN + n0;
  v8f acc0 = {}, acc1 = {};
#pragma unroll
  for (int k0 = 0; k0 < CHC; k0 += 32) {
    __syncthreads();
#pragma unroll
    for (int it = 0; it < 8; ++it) {
      int idx = tid + it * 256;
      int kk = idx >> 6, m = idx & 63;
      As[m][kk] = (_Float16)Qb[(size_t)(k0 + kk) * CHW + m];
    }
#pragma unroll
    for (int it = 0; it < 8; ++it) {
      int idx = tid + it * 256;
      int kk = idx >> 6, j = idx & 63;
      Bs[j][kk] = (_Float16)Kb[(size_t)(k0 + kk) * CN + j];
    }
    __syncthreads();
    v16h a  = frag_a_from_lds(&As[mi][0]);
    v16h b0 = frag_b_from_lds(&Bs[nj][0]);
    v16h b1 = frag_b_from_lds(&Bs[nj + 16][0]);
    acc0 = __builtin_amdgcn_wmma_f32_16x16x32_f16(false, a, false, b0, (short)0, acc0,
                                                  false, false);
    acc1 = __builtin_amdgcn_wmma_f32_16x16x32_f16(false, a, false, b1, (short)0, acc1,
                                                  false, false);
  }
  int lane = tid & 31;
  int half = lane >> 4, nn = lane & 15;
#pragma unroll
  for (int r = 0; r < 8; ++r) {
    int ml = blockIdx.x * 64 + mi + r + 8 * half;       // chunk-local row
    size_t rowb = ((size_t)bh * MCHUNK + ml) * CN + n0 + nj + nn;
    S[rowb]      = acc0[r];
    S[rowb + 16] = acc1[r];
  }
}

// ---------------- O[bh, c, m]: sum_j P[bh,ml,j]*v[bh,c,j] ---------------------------
// Block tile 64(M) x 64(C = all channels); K = 256 (8 steps).
__global__ __launch_bounds__(256)
void attn_out_kernel(const float* __restrict__ P, const float* __restrict__ V,
                     float* __restrict__ O, int m0chunk) {
  __shared__ _Float16 As[64][LDH];
  __shared__ _Float16 Bs[64][LDH];
  int tid = threadIdx.x;
  int wv  = tid >> 5;
  int mi  = (wv & 3) * 16;
  int nj  = (wv >> 2) * 32;
  int bh  = blockIdx.z;
  int mbase = blockIdx.x * 64;         // chunk-local
  const float* Pb = P + (size_t)(bh * MCHUNK + mbase) * CN;
  const float* Vb = V + (size_t)bh * CHC * CN;
  v8f acc0 = {}, acc1 = {};
  for (int k0 = 0; k0 < CN; k0 += 32) {
    __syncthreads();
    // A: 64 m x 32 j, vectorized along j (rows contiguous)
#pragma unroll
    for (int it = 0; it < 2; ++it) {
      int idx = tid + it * 256;
      int k4 = (idx & 7) * 4, m = idx >> 3;
      const float4 p4 = *(const float4*)&Pb[(size_t)m * CN + k0 + k4];
      As[m][k4 + 0] = (_Float16)p4.x;
      As[m][k4 + 1] = (_Float16)p4.y;
      As[m][k4 + 2] = (_Float16)p4.z;
      As[m][k4 + 3] = (_Float16)p4.w;
    }
    // B: 64 c x 32 j, vectorized along j
#pragma unroll
    for (int it = 0; it < 2; ++it) {
      int idx = tid + it * 256;
      int k4 = (idx & 7) * 4, c = idx >> 3;
      const float4 v4 = *(const float4*)&Vb[(size_t)c * CN + k0 + k4];
      Bs[c][k4 + 0] = (_Float16)v4.x;
      Bs[c][k4 + 1] = (_Float16)v4.y;
      Bs[c][k4 + 2] = (_Float16)v4.z;
      Bs[c][k4 + 3] = (_Float16)v4.w;
    }
    __syncthreads();
    v16h a  = frag_a_from_lds(&As[mi][0]);
    v16h b0 = frag_b_from_lds(&Bs[nj][0]);
    v16h b1 = frag_b_from_lds(&Bs[nj + 16][0]);
    acc0 = __builtin_amdgcn_wmma_f32_16x16x32_f16(false, a, false, b0, (short)0, acc0,
                                                  false, false);
    acc1 = __builtin_amdgcn_wmma_f32_16x16x32_f16(false, a, false, b1, (short)0, acc1,
                                                  false, false);
  }
  int lane = tid & 31;
  int half = lane >> 4, nn = lane & 15;
  int c0 = nj + nn, c1 = c0 + 16;
#pragma unroll
  for (int r = 0; r < 8; ++r) {
    int m = m0chunk + mbase + mi + r + 8 * half;
    O[((size_t)bh * CHC + c0) * CHW + m] = acc0[r];
    O[((size_t)bh * CHC + c1) * CHW + m] = acc1[r];
  }
}

// ---------------- offset branch: dwconv5x5 s2 -> LN(ch) -> GELU -> 1x1 -> tanh ------
__global__ __launch_bounds__(128)
void conv_offset_kernel(const float* __restrict__ Qbuf, const float* __restrict__ w_dw,
                        const float* __restrict__ b_dw, const float* __restrict__ ln_g,
                        const float* __restrict__ ln_b, const float* __restrict__ w_pw,
                        float* __restrict__ pos_out, float* __restrict__ ref_out) {
  int c = threadIdx.x;
  int sp = blockIdx.x;          // oy*16+ox
  int bg = blockIdx.y;
  int oy = sp >> 4, ox = sp & 15;
  int b = bg >> 2, g = bg & 3;
  const float* qc = Qbuf + (size_t)(b * CNC + g * CGC + c) * CHW;
  float acc = 0.0f;
#pragma unroll
  for (int i = 0; i < 5; ++i) {
    int iy = oy * 2 - 2 + i;
    if (iy < 0 || iy >= CH) continue;
#pragma unroll
    for (int j = 0; j < 5; ++j) {
      int ix = ox * 2 - 2 + j;
      if (ix < 0 || ix >= CW) continue;
      acc += qc[iy * CW + ix] * w_dw[c * 25 + i * 5 + j];
    }
  }
  acc += b_dw[c];
  __shared__ float red[128];
  __shared__ float red2[128];
  red[c] = acc;
  __syncthreads();
  for (int s = 64; s > 0; s >>= 1) { if (c < s) red[c] += red[c + s]; __syncthreads(); }
  float mu = red[0] * (1.0f / 128.0f);
  __syncthreads();
  float d = acc - mu;
  red[c] = d * d;
  __syncthreads();
  for (int s = 64; s > 0; s >>= 1) { if (c < s) red[c] += red[c + s]; __syncthreads(); }
  float var = red[0] * (1.0f / 128.0f);
  __syncthreads();
  float h = d * rsqrtf(var + 1e-5f) * ln_g[c] + ln_b[c];
  h = 0.5f * h * (1.0f + erff(h * 0.70710678118654752f));   // exact GELU
  red[c]  = w_pw[c] * h;            // output channel 0 (y)
  red2[c] = w_pw[128 + c] * h;      // output channel 1 (x)
  __syncthreads();
  for (int s = 64; s > 0; s >>= 1) {
    if (c < s) { red[c] += red[c + s]; red2[c] += red2[c + s]; }
    __syncthreads();
  }
  if (c == 0) {
    float offy = tanhf(red[0])  * (2.0f / 16.0f);
    float offx = tanhf(red2[0]) * (2.0f / 16.0f);
    // meshgrid 'xy' quirk: ref_y indexed by second coord, ref_x by first
    float refy = ((float)ox + 0.5f) / 16.0f * 2.0f - 1.0f;
    float refx = ((float)oy + 0.5f) / 16.0f * 2.0f - 1.0f;
    size_t o = ((size_t)bg * CN + sp) * 2;
    pos_out[o]     = offy + refy;
    pos_out[o + 1] = offx + refx;
    ref_out[o]     = refy;
    ref_out[o + 1] = refx;
  }
}

// ---------------- bilinear sample of x at pos -> x_s [B,512,256] --------------------
__global__ __launch_bounds__(128)
void sample_kernel(const float* __restrict__ X, const float* __restrict__ pos,
                   float* __restrict__ XS) {
  int c = threadIdx.x;
  int np = blockIdx.x;
  int bg = blockIdx.y;
  int b = bg >> 2, g = bg & 3;
  float py = pos[((size_t)bg * CN + np) * 2 + 0];
  float px = pos[((size_t)bg * CN + np) * 2 + 1];
  float gx = (px + 1.0f) * 0.5f * (float)(CW - 1);   // grid last dim flipped: x first
  float gy = (py + 1.0f) * 0.5f * (float)(CH - 1);
  float x0f = floorf(gx), y0f = floorf(gy);
  int x0 = (int)x0f, y0 = (int)y0f;
  float wx1 = gx - x0f, wy1 = gy - y0f;
  const float* xc = X + (size_t)(b * CNC + g * CGC + c) * CHW;
  float v00 = 0.f, v01 = 0.f, v10 = 0.f, v11 = 0.f;
  if (x0 >= 0 && x0 < CW && y0 >= 0 && y0 < CH)                 v00 = xc[y0 * CW + x0];
  if (x0 + 1 >= 0 && x0 + 1 < CW && y0 >= 0 && y0 < CH)         v01 = xc[y0 * CW + x0 + 1];
  if (x0 >= 0 && x0 < CW && y0 + 1 >= 0 && y0 + 1 < CH)         v10 = xc[(y0 + 1) * CW + x0];
  if (x0 + 1 >= 0 && x0 + 1 < CW && y0 + 1 >= 0 && y0 + 1 < CH) v11 = xc[(y0 + 1) * CW + x0 + 1];
  float val = v00 * (1.f - wy1) * (1.f - wx1) + v01 * (1.f - wy1) * wx1 +
              v10 * wy1 * (1.f - wx1) + v11 * wy1 * wx1;
  XS[(size_t)(b * CNC + g * CGC + c) * CN + np] = val;
}

// ---------------- RPE bias (bilinear over 63x63 table) + softmax --------------------
__global__ __launch_bounds__(256)
void bias_softmax_kernel(float* __restrict__ S, const float* __restrict__ pos,
                         const float* __restrict__ rpe, int m0chunk) {
  int j = threadIdx.x;
  int ml = blockIdx.x;
  int bh = blockIdx.y;
  int head = bh & 7;
  int bg = bh >> 1;                       // b*4 + g
  int m = m0chunk + ml;
  int qi = m >> 5, qj = m & 31;
  float qy = ((float)qj + 0.5f) / 32.0f * 2.0f - 1.0f;   // meshgrid 'xy' quirk
  float qx = ((float)qi + 0.5f) / 32.0f * 2.0f - 1.0f;
  float py = pos[((size_t)bg * CN + j) * 2 + 0];
  float px = pos[((size_t)bg * CN + j) * 2 + 1];
  float dy = (qy - py) * 0.5f;
  float dx = (qx - px) * 0.5f;
  float gx = (dx + 1.0f) * 0.5f * 62.0f;
  float gy = (dy + 1.0f) * 0.5f * 62.0f;
  float x0f = floorf(gx), y0f = floorf(gy);
  int x0 = (int)x0f, y0 = (int)y0f;
  float wx1 = gx - x0f, wy1 = gy - y0f;
  const float* tb = rpe + (size_t)head * 63 * 63;
  float v00 = 0.f, v01 = 0.f, v10 = 0.f, v11 = 0.f;
  if (x0 >= 0 && x0 < 63 && y0 >= 0 && y0 < 63)                 v00 = tb[y0 * 63 + x0];
  if (x0 + 1 >= 0 && x0 + 1 < 63 && y0 >= 0 && y0 < 63)         v01 = tb[y0 * 63 + x0 + 1];
  if (x0 >= 0 && x0 < 63 && y0 + 1 >= 0 && y0 + 1 < 63)         v10 = tb[(y0 + 1) * 63 + x0];
  if (x0 + 1 >= 0 && x0 + 1 < 63 && y0 + 1 >= 0 && y0 + 1 < 63) v11 = tb[(y0 + 1) * 63 + x0 + 1];
  float bias = v00 * (1.f - wy1) * (1.f - wx1) + v01 * (1.f - wy1) * wx1 +
               v10 * wy1 * (1.f - wx1) + v11 * wy1 * wx1;
  size_t idx = ((size_t)bh * MCHUNK + ml) * CN + j;
  float s = S[idx] * 0.125f + bias;        // SCALE = 64^-0.5
  __shared__ float red[256];
  red[j] = s;
  __syncthreads();
  for (int st = 128; st > 0; st >>= 1) {
    if (j < st) red[j] = fmaxf(red[j], red[j + st]);
    __syncthreads();
  }
  float mx = red[0];
  __syncthreads();
  float e = expf(s - mx);
  red[j] = e;
  __syncthreads();
  for (int st = 128; st > 0; st >>= 1) {
    if (j < st) red[j] += red[j + st];
    __syncthreads();
  }
  S[idx] = e / red[0];
}

// ---------------- launcher ----------------------------------------------------------
extern "C" void kernel_launch(void* const* d_in, const int* in_sizes, int n_in,
                              void* d_out, int out_size, void* d_ws, size_t ws_size,
                              hipStream_t stream) {
  (void)in_sizes; (void)n_in; (void)out_size; (void)ws_size;
  const float* x    = (const float*)d_in[0];
  const float* wq   = (const float*)d_in[1];
  const float* bq   = (const float*)d_in[2];
  const float* wk   = (const float*)d_in[3];
  const float* bk   = (const float*)d_in[4];
  const float* wv   = (const float*)d_in[5];
  const float* bv   = (const float*)d_in[6];
  const float* wo   = (const float*)d_in[7];
  const float* bo   = (const float*)d_in[8];
  const float* w_dw = (const float*)d_in[9];
  const float* b_dw = (const float*)d_in[10];
  const float* ln_g = (const float*)d_in[11];
  const float* ln_b = (const float*)d_in[12];
  const float* w_pw = (const float*)d_in[13];
  const float* rpe  = (const float*)d_in[14];

  float* out     = (float*)d_out;
  float* y_out   = out;
  float* pos_out = out + (size_t)CB * CNC * CHW;            // 8388608
  float* ref_out = pos_out + (size_t)CB * CG * CN * 2;      // +32768

  float* ws    = (float*)d_ws;
  float* q_ws  = ws;                                        // 16*512*1024
  float* xs_ws = q_ws + (size_t)CB * CNC * CHW;             // 16*512*256
  float* k_ws  = xs_ws + (size_t)CB * CNC * CN;
  float* v_ws  = k_ws + (size_t)CB * CNC * CN;
  float* s_ws  = v_ws + (size_t)CB * CNC * CN;              // 128*128*256
  float* o_ws  = s_ws + (size_t)CB * CNH * MCHUNK * CN;     // 16*512*1024

  // q projection
  gemm_proj_kernel<<<dim3(256, 8), 256, 0, stream>>>(x, wq, bq, q_ws, CHW, CNC, CNC);
  // offsets -> pos / ref (written straight into d_out)
  conv_offset_kernel<<<dim3(256, 64), 128, 0, stream>>>(q_ws, w_dw, b_dw, ln_g, ln_b,
                                                        w_pw, pos_out, ref_out);
  // deformable sampling
  sample_kernel<<<dim3(256, 64), 128, 0, stream>>>(x, pos_out, xs_ws);
  // k / v projections
  gemm_proj_kernel<<<dim3(64, 8), 256, 0, stream>>>(xs_ws, wk, bk, k_ws, CN, CNC, CNC);
  gemm_proj_kernel<<<dim3(64, 8), 256, 0, stream>>>(xs_ws, wv, bv, v_ws, CN, CNC, CNC);
  // attention, chunked over query rows
  for (int mc = 0; mc < CHW / MCHUNK; ++mc) {
    int m0 = mc * MCHUNK;
    attn_score_kernel<<<dim3(2, 4, 128), 256, 0, stream>>>(q_ws, k_ws, s_ws, m0);
    bias_softmax_kernel<<<dim3(128, 128), 256, 0, stream>>>(s_ws, pos_out, rpe, m0);
    attn_out_kernel<<<dim3(2, 1, 128), 256, 0, stream>>>(s_ws, v_ws, o_ws, m0);
  }
  // output projection
  gemm_proj_kernel<<<dim3(256, 8), 256, 0, stream>>>(o_ws, wo, bo, y_out, CHW, CNC, CNC);
}
